// Encoder_39281770889454
// MI455X (gfx1250) — matrix operand order repaired and checked by reading, not compiled, not run on hardware
//
#include <hip/hip_runtime.h>

typedef __attribute__((ext_vector_type(16))) __bf16 v16bf;
typedef __attribute__((ext_vector_type(8)))  float  v8f;
typedef __attribute__((ext_vector_type(4)))  float  v4f;

// ---------- degree / norm ----------
__global__ void deg_init_kernel(float* __restrict__ deg, int N) {
  int i = blockIdx.x * blockDim.x + threadIdx.x;
  if (i < N) deg[i] = 1.0f;  // self loop contributes 1
}

__global__ void deg_accum_kernel(const long long* __restrict__ ei,
                                 float* __restrict__ deg, int E) {
  int e = blockIdx.x * blockDim.x + threadIdx.x;
  if (e < E) {
    int d = (int)ei[E + e];  // row 1 = dst
    atomicAdd(&deg[d], 1.0f);
  }
}

__global__ void rsqrt_inplace_kernel(float* __restrict__ v, int N) {
  int i = blockIdx.x * blockDim.x + threadIdx.x;
  if (i < N) v[i] = rsqrtf(v[i]);  // deg >= 1 always (self loop)
}

__global__ void edge_prep_kernel(const long long* __restrict__ ei,
                                 const float* __restrict__ dinv,
                                 int* __restrict__ src32, int* __restrict__ dst32,
                                 float* __restrict__ norm, int E) {
  int e = blockIdx.x * blockDim.x + threadIdx.x;
  if (e < E) {
    int s = (int)ei[e];
    int d = (int)ei[E + e];
    src32[e] = s;
    dst32[e] = d;
    norm[e] = dinv[s] * dinv[d];
  }
}

// ---------- WMMA GEMM: H[M x 64] = X[M x K] * W[K x 64] ----------
// K multiple of 32 (256 or 64), M multiple of 16.
// Block = 128 threads = 4 waves; each wave computes a 32x64 output tile
// (two 16-row A fragments sharing every B fragment -> 2 WMMAs per B pair of
// ds_load_b128). Tail 16-row tile handled by the FULL=false instantiation,
// keeping all branches wave-uniform so EXEC is all-ones at every WMMA.

template <bool FULL>
static __device__ __forceinline__ void gemm_tile(
    const float* __restrict__ X, const __bf16* sB, float* __restrict__ H,
    int K, int row0, int lane) {
  const int laneLo = lane & 15;   // row within A tile / col within B tile
  const int laneHi = lane >> 4;   // K-half selector

  const float* __restrict__ xr0 =
      X + (long long)(row0 + laneLo) * K + laneHi * 8;
  const float* __restrict__ xr1 =
      X + (long long)(row0 + 16 + laneLo) * K + laneHi * 8;

  const int KS = K >> 5;
  v8f acc[8] = {};

  for (int ks = 0; ks < KS; ++ks) {
    // A fragment (16x32 bf16, ISA 7.12.2): per lane, K runs {c..c+7} and
    // {c+16..c+23} with c = laneHi*8 -> four aligned b128 loads + native cvt.
    const float* xa0 = xr0 + ks * 32;
    v4f p0 = *(const v4f*)(xa0);
    v4f p1 = *(const v4f*)(xa0 + 4);
    v4f p2 = *(const v4f*)(xa0 + 16);
    v4f p3 = *(const v4f*)(xa0 + 20);
    __builtin_prefetch(xa0 + 32, 0, 0);  // global_prefetch_b8, streaming A

    v16bf a0;
#pragma unroll
    for (int j = 0; j < 4; ++j) {
      a0[j]      = (__bf16)p0[j];
      a0[4 + j]  = (__bf16)p1[j];
      a0[8 + j]  = (__bf16)p2[j];
      a0[12 + j] = (__bf16)p3[j];
    }

    v16bf a1;
    if constexpr (FULL) {
      const float* xa1 = xr1 + ks * 32;
      v4f q0 = *(const v4f*)(xa1);
      v4f q1 = *(const v4f*)(xa1 + 4);
      v4f q2 = *(const v4f*)(xa1 + 16);
      v4f q3 = *(const v4f*)(xa1 + 20);
      __builtin_prefetch(xa1 + 32, 0, 0);
#pragma unroll
      for (int j = 0; j < 4; ++j) {
        a1[j]      = (__bf16)q0[j];
        a1[4 + j]  = (__bf16)q1[j];
        a1[8 + j]  = (__bf16)q2[j];
        a1[12 + j] = (__bf16)q3[j];
      }
    }

#pragma unroll
    for (int t = 0; t < 4; ++t) {
      const v16bf b =
          *(const v16bf*)&sB[(((ks * 4 + t) * 32) + lane) * 16];
      acc[t] = __builtin_amdgcn_wmma_f32_16x16x32_bf16(
          false, a0, false, b, (short)0, acc[t], false, false);
      if constexpr (FULL) {
        acc[4 + t] = __builtin_amdgcn_wmma_f32_16x16x32_bf16(
            false, a1, false, b, (short)0, acc[4 + t], false, false);
      }
    }
  }

  // D layout: VGPR r -> row = row0 + r + 8*laneHi, col = t*16 + laneLo
#pragma unroll
  for (int t = 0; t < 4; ++t) {
    const int col = t * 16 + laneLo;
#pragma unroll
    for (int r = 0; r < 8; ++r) {
      H[(long long)(row0 + r + laneHi * 8) * 64 + col] = acc[t][r];
      if constexpr (FULL) {
        H[(long long)(row0 + 16 + r + laneHi * 8) * 64 + col] = acc[4 + t][r];
      }
    }
  }
}

__global__ void __launch_bounds__(128)
gcn_gemm_wmma(const float* __restrict__ X, const float* __restrict__ W,
              float* __restrict__ H, int M, int K) {
  // packed B-fragment layout: sB[(((ks*4 + t)*32 + lane)*16) + h]
  //   value = bf16( W[(ks*32 + (h&7) + (lane>>4)*8 + (h>>3)*16) * 64 + (t*16 + (lane&15))] )
  __shared__ __bf16 sB[256 * 64];  // max K=256 -> 32 KB (<< 320 KB/WGP)

  const int tid = threadIdx.x;
  for (int p = tid; p < K * 64; p += 128) {
    const int h    = p & 15;
    const int lane = (p >> 4) & 31;
    const int t    = (p >> 9) & 3;
    const int s    = p >> 11;
    const int kk   = s * 32 + (h & 7) + (lane >> 4) * 8 + (h >> 3) * 16;
    const int col  = t * 16 + (lane & 15);
    sB[p] = (__bf16)W[kk * 64 + col];
  }
  __syncthreads();

  const int lane       = tid & 31;
  const int waveGlobal = blockIdx.x * 4 + (tid >> 5);
  const int row0       = waveGlobal * 32;
  if (row0 >= M) return;          // wave-uniform

  if (row0 + 32 <= M)
    gemm_tile<true>(X, sB, H, K, row0, lane);
  else
    gemm_tile<false>(X, sB, H, K, row0, lane);  // 16-row tail tile
}

// ---------- aggregation (float4-vectorized) ----------
__global__ void self_init_kernel(const float* __restrict__ H,
                                 const float* __restrict__ dinv,
                                 float* __restrict__ agg, int total4, int OUT) {
  int idx = blockIdx.x * blockDim.x + threadIdx.x;
  if (idx < total4) {
    const int base = idx * 4;
    const int i = base / OUT;
    const float di = dinv[i];
    const float c = di * di;
    v4f h = *(const v4f*)(H + base);
    v4f r = {h.x * c, h.y * c, h.z * c, h.w * c};
    *(v4f*)(agg + base) = r;  // self-loop message, also initializes agg
  }
}

__global__ void edge_scatter_kernel(const float* __restrict__ H,
                                    const int* __restrict__ src32,
                                    const int* __restrict__ dst32,
                                    const float* __restrict__ norm,
                                    float* __restrict__ agg, int E, int OUT) {
  int idx = blockIdx.x * blockDim.x + threadIdx.x;
  const int vecs = OUT >> 2;           // 16 float4s per edge
  int e  = idx / vecs;
  int c4 = (idx - e * vecs) * 4;
  if (e < E) {
    const float w = norm[e];
    const long long sb = (long long)src32[e] * OUT + c4;
    const long long db = (long long)dst32[e] * OUT + c4;
    v4f h = *(const v4f*)(H + sb);
    atomicAdd(&agg[db + 0], h.x * w);
    atomicAdd(&agg[db + 1], h.y * w);
    atomicAdd(&agg[db + 2], h.z * w);
    atomicAdd(&agg[db + 3], h.w * w);
  }
}

__global__ void bias_relu_kernel(const float* __restrict__ A,
                                 const float* __restrict__ b,
                                 float* __restrict__ Z, int total4, int OUT) {
  int idx = blockIdx.x * blockDim.x + threadIdx.x;
  if (idx < total4) {
    const int base = idx * 4;
    const int col = base & (OUT - 1);  // OUT = 64, power of two
    v4f a = *(const v4f*)(A + base);
    v4f bb = *(const v4f*)(b + col);
    v4f r = {fmaxf(a.x + bb.x, 0.0f), fmaxf(a.y + bb.y, 0.0f),
             fmaxf(a.z + bb.z, 0.0f), fmaxf(a.w + bb.w, 0.0f)};
    *(v4f*)(Z + base) = r;
  }
}

// ---------- launcher ----------
extern "C" void kernel_launch(void* const* d_in, const int* in_sizes, int n_in,
                              void* d_out, int out_size, void* d_ws, size_t ws_size,
                              hipStream_t stream) {
  const float* feat   = (const float*)d_in[0];
  const float* W1     = (const float*)d_in[1];
  const float* b1     = (const float*)d_in[2];
  const float* W2     = (const float*)d_in[3];
  const float* b2     = (const float*)d_in[4];
  const long long* ei = (const long long*)d_in[5];  // int64 [2, E]

  const int OUT = in_sizes[2];           // 64
  const int IN  = in_sizes[1] / OUT;     // 256
  const int N   = in_sizes[0] / IN;      // 50000
  const int E   = in_sizes[5] / 2;       // 800000

  float* out = (float*)d_out;

  // workspace layout (all offsets 16B-aligned)
  float* dinv  = (float*)d_ws;           // N   (holds deg, then rsqrt in place)
  float* norm  = dinv + N;               // E
  int*   src32 = (int*)(norm + E);       // E
  int*   dst32 = src32 + E;              // E
  float* bufA  = (float*)(dst32 + E);    // N*OUT  (H1, then Z1)
  float* bufB  = bufA + (size_t)N * OUT; // N*OUT  (AGG1, then H2)

  const int tpb = 256;
  const unsigned gN   = (N + tpb - 1) / tpb;
  const unsigned gE   = (E + tpb - 1) / tpb;
  const int      no4  = (N * OUT) / 4;
  const unsigned gNO4 = ((unsigned)no4 + tpb - 1) / tpb;
  const unsigned gEO4 = (unsigned)(((long long)E * (OUT / 4) + tpb - 1) / tpb);

  // normalization
  deg_init_kernel<<<gN, tpb, 0, stream>>>(dinv, N);
  deg_accum_kernel<<<gE, tpb, 0, stream>>>(ei, dinv, E);
  rsqrt_inplace_kernel<<<gN, tpb, 0, stream>>>(dinv, N);
  edge_prep_kernel<<<gE, tpb, 0, stream>>>(ei, dinv, src32, dst32, norm, E);

  const unsigned tiles32    = (unsigned)((N + 31) / 32);
  const unsigned gemmBlocks = (tiles32 + 3) / 4;

  // layer 1: H1 = feat @ W1 ; agg ; z1 = relu(agg + b1)
  gcn_gemm_wmma<<<gemmBlocks, 128, 0, stream>>>(feat, W1, bufA, N, IN);
  self_init_kernel<<<gNO4, tpb, 0, stream>>>(bufA, dinv, bufB, no4, OUT);
  edge_scatter_kernel<<<gEO4, tpb, 0, stream>>>(bufA, src32, dst32, norm, bufB, E, OUT);
  bias_relu_kernel<<<gNO4, tpb, 0, stream>>>(bufB, b1, bufA, no4, OUT);

  // layer 2: H2 = z1 @ W2 ; agg into d_out ; relu(+b2) in place
  gcn_gemm_wmma<<<gemmBlocks, 128, 0, stream>>>(bufA, W2, bufB, N, OUT);
  self_init_kernel<<<gNO4, tpb, 0, stream>>>(bufB, dinv, out, no4, OUT);
  edge_scatter_kernel<<<gEO4, tpb, 0, stream>>>(bufB, src32, dst32, norm, out, E, OUT);
  bias_relu_kernel<<<gNO4, tpb, 0, stream>>>(out, b2, out, N * OUT / 4, OUT);
}
